// SchNet_14697378087519
// MI455X (gfx1250) — matrix operand order repaired and checked by reading, not compiled, not run on hardware
//
#include <hip/hip_runtime.h>
#include <hip/hip_bf16.h>

#define N_NODES 8192
#define N_EDGES 262144
#define HID 128
#define NRBF 50
#define KRBF 64          // padded K for rbf
#define NLAYERS 6
#define PI_F 3.14159265358979323846f

typedef __bf16 bf16_t;
typedef bf16_t v16bf __attribute__((ext_vector_type(16)));
typedef bf16_t v8bf  __attribute__((ext_vector_type(8)));
typedef float  v8f   __attribute__((ext_vector_type(8)));
typedef int    v4i_t __attribute__((vector_size(16)));   // matches builtin param type

// Detect the gfx1250 async global->LDS path (ASYNCcnt tracked). Fallback: plain copy.
#if defined(__has_builtin)
#if __has_builtin(__builtin_amdgcn_global_load_async_to_lds_b128) && \
    __has_builtin(__builtin_amdgcn_s_wait_asynccnt)
#define USE_ASYNC_LDS 1
#endif
#endif

#define AS1P __attribute__((address_space(1))) v4i_t*
#define AS3P __attribute__((address_space(3))) v4i_t*

// ---------------------------------------------------------------- WMMA helpers

__device__ __forceinline__ v8f wmma_bf16(v16bf a, v16bf b, v8f c) {
  return __builtin_amdgcn_wmma_f32_16x16x32_bf16(
      /*neg_a=*/false, a, /*neg_b=*/false, b,
      /*c_mod=*/(short)0, c, /*reuse_a=*/false, /*reuse_b=*/false);
}

// A-fragment of a 16x32 bf16 tile from row-major [rows][ld] bf16 storage.
// lane row = L&15; khalf = 8*(L>=16). Regs 0-3: K kh..kh+7, regs 4-7: K kh+16..kh+23.
__device__ __forceinline__ v16bf load_a_bf(const bf16_t* __restrict__ A, int ld,
                                           int m, int kbase, int khalf) {
  const bf16_t* p = A + (size_t)m * ld + kbase + khalf;
  v8bf lo = *reinterpret_cast<const v8bf*>(p);
  v8bf hi = *reinterpret_cast<const v8bf*>(p + 16);
  v16bf a;
#pragma unroll
  for (int i = 0; i < 8; ++i) { a[i] = lo[i]; a[8 + i] = hi[i]; }
  return a;
}

// Same A-fragment but converting from fp32 row-major activations on the fly.
__device__ __forceinline__ v16bf load_a_f32(const float* __restrict__ A, int ld,
                                            int m, int kbase, int khalf) {
  const float* p = A + (size_t)m * ld + kbase + khalf;
  float4 f0 = *reinterpret_cast<const float4*>(p);
  float4 f1 = *reinterpret_cast<const float4*>(p + 4);
  float4 f2 = *reinterpret_cast<const float4*>(p + 16);
  float4 f3 = *reinterpret_cast<const float4*>(p + 20);
  v16bf a;
  a[0]=(bf16_t)f0.x; a[1]=(bf16_t)f0.y; a[2]=(bf16_t)f0.z; a[3]=(bf16_t)f0.w;
  a[4]=(bf16_t)f1.x; a[5]=(bf16_t)f1.y; a[6]=(bf16_t)f1.z; a[7]=(bf16_t)f1.w;
  a[8]=(bf16_t)f2.x; a[9]=(bf16_t)f2.y; a[10]=(bf16_t)f2.z; a[11]=(bf16_t)f2.w;
  a[12]=(bf16_t)f3.x; a[13]=(bf16_t)f3.y; a[14]=(bf16_t)f3.z; a[15]=(bf16_t)f3.w;
  return a;
}

// B-fragment of a 32x16 bf16 tile from transposed weights WT[col][k] (ldk = K).
// lane column = L&15; khalf16 = 16*(L>=16); regs 0-3: K kh..kh+7, regs 4-7: K kh+16..kh+23 (kh = khalf16/2).
// Works for both global and LDS-resident WT.
__device__ __forceinline__ v16bf load_b_bf(const bf16_t* WT, int ldk,
                                           int col, int kbase, int khalf16) {
  const bf16_t* p = WT + (size_t)col * ldk + kbase + khalf16 / 2;
  v8bf lo = *reinterpret_cast<const v8bf*>(p);
  v8bf hi = *reinterpret_cast<const v8bf*>(p + 16);
  v16bf b;
#pragma unroll
  for (int i = 0; i < 8; ++i) { b[i] = lo[i]; b[8 + i] = hi[i]; }
  return b;
}

// Cooperative block staging of bf16 weight slabs into LDS (16B granules).
__device__ __forceinline__ void stage_to_lds(bf16_t* dst, const bf16_t* __restrict__ src,
                                             int nhalf, int tid, int nthreads) {
  int nchunks = nhalf / 8;
#ifdef USE_ASYNC_LDS
  for (int i = tid; i < nchunks; i += nthreads)
    __builtin_amdgcn_global_load_async_to_lds_b128(
        (AS1P)(const void*)(src + (size_t)i * 8),
        (AS3P)(void*)(dst + (size_t)i * 8),
        /*offset=*/0, /*cpol=*/0);
#else
  for (int i = tid; i < nchunks; i += nthreads)
    reinterpret_cast<v8bf*>(dst)[i] = reinterpret_cast<const v8bf*>(src)[i];
#endif
}

__device__ __forceinline__ void stage_wait() {
#ifdef USE_ASYNC_LDS
  __builtin_amdgcn_s_wait_asynccnt(0);
#endif
  __syncthreads();
}

__device__ __forceinline__ float silu_f(float v) { return v / (1.0f + __expf(-v)); }

// ---------------------------------------------------------------- prep kernels

// fp32 [K][Nm] -> bf16 transposed [Nm][Kp] (zero padded K)
__global__ void transpose_bf16_kernel(const float* __restrict__ W, bf16_t* __restrict__ WT,
                                      int K, int Nm, int Kp) {
  int idx = blockIdx.x * blockDim.x + threadIdx.x;
  if (idx >= Nm * Kp) return;
  int n = idx / Kp, k = idx % Kp;
  WT[idx] = (k < K) ? (bf16_t)W[(size_t)k * Nm + n] : (bf16_t)0.0f;
}

// per-edge cutoff + ExpNormalSmearing -> bf16 [E][64] (padded) + cut [E]
__global__ void rbf_kernel(const float* __restrict__ ew, const float* __restrict__ means,
                           const float* __restrict__ betas, float* __restrict__ cut,
                           bf16_t* __restrict__ rbfb) {
  int e = blockIdx.x * blockDim.x + threadIdx.x;
  if (e >= N_EDGES) return;
  float d = ew[e];
  float c = (d < 5.0f) ? 0.5f * (__cosf(d * (PI_F / 5.0f)) + 1.0f) : 0.0f;
  cut[e] = c;
  float t = __expf(-d);                  // exp(ALPHA*(START-d)), ALPHA=1, START=0
  bf16_t* o = rbfb + (size_t)e * KRBF;
#pragma unroll 2
  for (int i = 0; i < NRBF; ++i) {
    float x = t - means[i];
    o[i] = (bf16_t)(c * __expf(-betas[i] * x * x));
  }
#pragma unroll
  for (int i = NRBF; i < KRBF; ++i) o[i] = (bf16_t)0.0f;
}

// x0 = emb[z], y = ne_emb[z]
__global__ void embed_kernel(const int* __restrict__ z, const float* __restrict__ emb,
                             const float* __restrict__ nemb, float* __restrict__ x0,
                             float* __restrict__ y) {
  int idx = blockIdx.x * blockDim.x + threadIdx.x;    // N*32 threads, float4 each
  if (idx >= N_NODES * (HID / 4)) return;
  unsigned n = (unsigned)idx >> 5, q = (unsigned)idx & 31u;
  unsigned zi = (unsigned)z[n];
  reinterpret_cast<float4*>(x0)[n * 32u + q] =
      reinterpret_cast<const float4*>(emb)[zi * 32u + q];
  reinterpret_cast<float4*>(y)[n * 32u + q] =
      reinterpret_cast<const float4*>(nemb)[zi * 32u + q];
}

// ---------------------------------------------------------------- edge kernels

// NeighborEmbedding: W = (rbf@projW + b)*(cut*mask); atomicAdd xn[col] += y[row]*W
__global__ __launch_bounds__(128) void neighbor_edge_kernel(
    const bf16_t* __restrict__ rbfb, const float* __restrict__ cut,
    const int* __restrict__ row, const int* __restrict__ col,
    const float* __restrict__ y, const bf16_t* __restrict__ projT,
    const float* __restrict__ projB, float* __restrict__ xn) {
  __shared__ alignas(16) bf16_t sW[HID * KRBF];       // 16 KB
  int tid = threadIdx.x;
  int lane = tid & 31, wv = tid >> 5;
  int tile = blockIdx.x * 4 + wv;
  int e0 = tile * 16;
  int m = lane & 15, hi = lane >> 4;
  int khA = hi * 8, khB = hi * 16;

  stage_to_lds(sW, projT, HID * KRBF, tid, 128);

  const bf16_t* Arow = rbfb + (size_t)e0 * KRBF;
  v16bf a0 = load_a_bf(Arow, KRBF, m, 0, khA);
  v16bf a1 = load_a_bf(Arow, KRBF, m, 32, khA);

  unsigned rowOff[8], colOff[8]; float sv[8];
#pragma unroll
  for (int r = 0; r < 8; ++r) {
    int e = e0 + r + hi * 8;
    unsigned rr = (unsigned)row[e], cc = (unsigned)col[e];
    rowOff[r] = rr * (unsigned)HID;
    colOff[r] = cc * (unsigned)HID;
    sv[r] = (rr != cc) ? cut[e] : 0.0f;               // self-loop mask
    __builtin_prefetch(y + rowOff[r], 0, 0);
  }
  stage_wait();

#pragma unroll
  for (int t = 0; t < 8; ++t) {
    int ncol = t * 16 + m;
    v8f c = {0.f, 0.f, 0.f, 0.f, 0.f, 0.f, 0.f, 0.f};
    c = wmma_bf16(a0, load_b_bf(sW, KRBF, ncol, 0, khB), c);
    c = wmma_bf16(a1, load_b_bf(sW, KRBF, ncol, 32, khB), c);
    float bias = projB[ncol];
#pragma unroll
    for (int r = 0; r < 8; ++r) {
      float w = (c[r] + bias) * sv[r];
      float msg = w * y[rowOff[r] + (unsigned)ncol];
      atomicAdd(&xn[colOff[r] + (unsigned)ncol], msg);
    }
  }
}

// CFConv: Wf = (silu(rbf@W1+b1)@W2+b2)*cut; atomicAdd agg[col] += h[row]*Wf
__global__ __launch_bounds__(128) void cfconv_edge_kernel(
    const bf16_t* __restrict__ rbfb, const float* __restrict__ cut,
    const int* __restrict__ row, const int* __restrict__ col,
    const float* __restrict__ h, const bf16_t* __restrict__ w1T,
    const float* __restrict__ b1, const bf16_t* __restrict__ w2T,
    const float* __restrict__ b2, float* __restrict__ agg) {
  __shared__ alignas(16) bf16_t sW1[HID * KRBF];      // 16 KB
  __shared__ alignas(16) bf16_t sW2[HID * HID];       // 32 KB
  __shared__ alignas(16) bf16_t sTile[4 * 16 * HID];  // 16 KB (per-wave silu tiles)
  int tid = threadIdx.x;
  int lane = tid & 31, wv = tid >> 5;
  int tile = blockIdx.x * 4 + wv;
  int e0 = tile * 16;
  int m = lane & 15, hi = lane >> 4;
  int khA = hi * 8, khB = hi * 16;
  bf16_t* lb = sTile + wv * 16 * HID;

  stage_to_lds(sW1, w1T, HID * KRBF, tid, 128);
  stage_to_lds(sW2, w2T, HID * HID, tid, 128);

  const bf16_t* Arow = rbfb + (size_t)e0 * KRBF;
  v16bf a0 = load_a_bf(Arow, KRBF, m, 0, khA);
  v16bf a1 = load_a_bf(Arow, KRBF, m, 32, khA);

  unsigned rowOff[8], colOff[8]; float sv[8];
#pragma unroll
  for (int r = 0; r < 8; ++r) {
    int e = e0 + r + hi * 8;
    unsigned rr = (unsigned)row[e], cc = (unsigned)col[e];
    rowOff[r] = rr * (unsigned)HID;
    colOff[r] = cc * (unsigned)HID;
    sv[r] = cut[e];
    __builtin_prefetch(h + rowOff[r], 0, 0);          // global_prefetch gathered rows
  }
  stage_wait();

  // GEMM1: [16x64]@[64x128] + bias, silu, -> LDS bf16 tile (C-layout store)
#pragma unroll
  for (int t = 0; t < 8; ++t) {
    int ncol = t * 16 + m;
    v8f c = {0.f, 0.f, 0.f, 0.f, 0.f, 0.f, 0.f, 0.f};
    c = wmma_bf16(a0, load_b_bf(sW1, KRBF, ncol, 0, khB), c);
    c = wmma_bf16(a1, load_b_bf(sW1, KRBF, ncol, 32, khB), c);
    float bias = b1[ncol];
#pragma unroll
    for (int r = 0; r < 8; ++r)
      lb[(r + hi * 8) * HID + ncol] = (bf16_t)silu_f(c[r] + bias);
  }
  __syncthreads();

  // re-load LDS tile in A-layout for GEMM2
  v16bf A2[4];
#pragma unroll
  for (int kc = 0; kc < 4; ++kc) A2[kc] = load_a_bf(lb, HID, m, kc * 32, khA);

  // GEMM2 [16x128]@[128x128] + bias, *cut, fused gather-multiply-scatter epilogue
#pragma unroll
  for (int t = 0; t < 8; ++t) {
    int ncol = t * 16 + m;
    v8f c = {0.f, 0.f, 0.f, 0.f, 0.f, 0.f, 0.f, 0.f};
#pragma unroll
    for (int kc = 0; kc < 4; ++kc)
      c = wmma_bf16(A2[kc], load_b_bf(sW2, HID, ncol, kc * 32, khB), c);
    float bias = b2[ncol];
#pragma unroll
    for (int r = 0; r < 8; ++r) {
      float wf = (c[r] + bias) * sv[r];
      float msg = wf * h[rowOff[r] + (unsigned)ncol];
      atomicAdd(&agg[colOff[r] + (unsigned)ncol], msg);
    }
  }
}

// ---------------------------------------------------------------- node kernels

// h = X @ lin1T  (no bias)
__global__ __launch_bounds__(128) void lin1_kernel(const float* __restrict__ X,
                                                   const bf16_t* __restrict__ wT,
                                                   float* __restrict__ hout) {
  int lane = threadIdx.x & 31, wv = threadIdx.x >> 5;
  int n0 = (blockIdx.x * 4 + wv) * 16;
  int m = lane & 15, hi = lane >> 4;
  int khA = hi * 8, khB = hi * 16;
  v16bf A[4];
#pragma unroll
  for (int kc = 0; kc < 4; ++kc)
    A[kc] = load_a_f32(X + (size_t)n0 * HID, HID, m, kc * 32, khA);
#pragma unroll
  for (int t = 0; t < 8; ++t) {
    int ncol = t * 16 + m;
    v8f c = {0.f, 0.f, 0.f, 0.f, 0.f, 0.f, 0.f, 0.f};
#pragma unroll
    for (int kc = 0; kc < 4; ++kc)
      c = wmma_bf16(A[kc], load_b_bf(wT, HID, ncol, kc * 32, khB), c);
#pragma unroll
    for (int r = 0; r < 8; ++r)
      hout[(unsigned)((n0 + r + hi * 8) * HID + ncol)] = c[r];
  }
}

// X = concat(x0, xn) @ catT + bias   ([N,256]@[256,128])
__global__ __launch_bounds__(128) void cat_kernel(const float* __restrict__ x0,
                                                  const float* __restrict__ xn,
                                                  const bf16_t* __restrict__ wT,
                                                  const float* __restrict__ bias,
                                                  float* __restrict__ X) {
  int lane = threadIdx.x & 31, wv = threadIdx.x >> 5;
  int n0 = (blockIdx.x * 4 + wv) * 16;
  int m = lane & 15, hi = lane >> 4;
  int khA = hi * 8, khB = hi * 16;
  v16bf A[8];
#pragma unroll
  for (int kc = 0; kc < 8; ++kc) {
    const float* src = (kc < 4) ? x0 : xn;
    A[kc] = load_a_f32(src + (size_t)n0 * HID, HID, m, (kc & 3) * 32, khA);
  }
#pragma unroll
  for (int t = 0; t < 8; ++t) {
    int ncol = t * 16 + m;
    v8f c = {0.f, 0.f, 0.f, 0.f, 0.f, 0.f, 0.f, 0.f};
#pragma unroll
    for (int kc = 0; kc < 8; ++kc)
      c = wmma_bf16(A[kc], load_b_bf(wT, 2 * HID, ncol, kc * 32, khB), c);
    float b = bias[ncol];
#pragma unroll
    for (int r = 0; r < 8; ++r)
      X[(unsigned)((n0 + r + hi * 8) * HID + ncol)] = c[r] + b;
  }
}

// X += silu(agg @ lin2T + b2) @ linT + bO   (fused via per-wave LDS tile)
__global__ __launch_bounds__(128) void node_update_kernel(
    const float* __restrict__ agg, const bf16_t* __restrict__ w2T,
    const float* __restrict__ b2, const bf16_t* __restrict__ wT,
    const float* __restrict__ bO, float* __restrict__ X) {
  __shared__ alignas(16) bf16_t sm[4 * 16 * HID];
  int lane = threadIdx.x & 31, wv = threadIdx.x >> 5;
  int n0 = (blockIdx.x * 4 + wv) * 16;
  int m = lane & 15, hi = lane >> 4;
  int khA = hi * 8, khB = hi * 16;
  bf16_t* lb = sm + wv * 16 * HID;

  v16bf A[4];
#pragma unroll
  for (int kc = 0; kc < 4; ++kc)
    A[kc] = load_a_f32(agg + (size_t)n0 * HID, HID, m, kc * 32, khA);
#pragma unroll
  for (int t = 0; t < 8; ++t) {
    int ncol = t * 16 + m;
    v8f c = {0.f, 0.f, 0.f, 0.f, 0.f, 0.f, 0.f, 0.f};
#pragma unroll
    for (int kc = 0; kc < 4; ++kc)
      c = wmma_bf16(A[kc], load_b_bf(w2T, HID, ncol, kc * 32, khB), c);
    float b = b2[ncol];
#pragma unroll
    for (int r = 0; r < 8; ++r)
      lb[(r + hi * 8) * HID + ncol] = (bf16_t)silu_f(c[r] + b);
  }
  __syncthreads();

  v16bf A2[4];
#pragma unroll
  for (int kc = 0; kc < 4; ++kc) A2[kc] = load_a_bf(lb, HID, m, kc * 32, khA);
#pragma unroll
  for (int t = 0; t < 8; ++t) {
    int ncol = t * 16 + m;
    v8f c = {0.f, 0.f, 0.f, 0.f, 0.f, 0.f, 0.f, 0.f};
#pragma unroll
    for (int kc = 0; kc < 4; ++kc)
      c = wmma_bf16(A2[kc], load_b_bf(wT, HID, ncol, kc * 32, khB), c);
    float b = bO[ncol];
#pragma unroll
    for (int r = 0; r < 8; ++r)
      X[(unsigned)((n0 + r + hi * 8) * HID + ncol)] += c[r] + b;
  }
}

// ---------------------------------------------------------------- host driver

extern "C" void kernel_launch(void* const* d_in, const int* in_sizes, int n_in,
                              void* d_out, int out_size, void* d_ws, size_t ws_size,
                              hipStream_t stream) {
  const int*   z         = (const int*)  d_in[0];
  const int*   row       = (const int*)  d_in[1];            // edge_index[0]
  const int*   col       = ((const int*) d_in[1]) + N_EDGES; // edge_index[1]
  const float* ew        = (const float*)d_in[2];
  const float* emb       = (const float*)d_in[3];
  const float* ne_emb    = (const float*)d_in[4];
  const float* ne_proj_w = (const float*)d_in[5];
  const float* ne_proj_b = (const float*)d_in[6];
  const float* ne_cat_w  = (const float*)d_in[7];
  const float* ne_cat_b  = (const float*)d_in[8];
  const float* rbf_means = (const float*)d_in[9];
  const float* rbf_betas = (const float*)d_in[10];
  const float* mlp_w1    = (const float*)d_in[11];
  const float* mlp_b1    = (const float*)d_in[12];
  const float* mlp_w2    = (const float*)d_in[13];
  const float* mlp_b2    = (const float*)d_in[14];
  const float* lin1_w    = (const float*)d_in[15];
  const float* lin2_w    = (const float*)d_in[16];
  const float* lin2_b    = (const float*)d_in[17];
  const float* lin_w     = (const float*)d_in[18];
  const float* lin_b     = (const float*)d_in[19];
  float* X = (float*)d_out;   // [N,HID] output buffer, used as the live x

  // ---- workspace layout
  char* ws = (char*)d_ws;
  size_t off = 0;
  auto take = [&](size_t bytes) -> void* {
    off = (off + 255) & ~(size_t)255;
    void* p = ws + off; off += bytes; return p;
  };
  float*  cut  = (float*) take((size_t)N_EDGES * 4);
  bf16_t* rbfb = (bf16_t*)take((size_t)N_EDGES * KRBF * 2);
  float*  x0   = (float*) take((size_t)N_NODES * HID * 4);
  float*  y    = (float*) take((size_t)N_NODES * HID * 4);
  float*  xn   = (float*) take((size_t)N_NODES * HID * 4);
  float*  h    = (float*) take((size_t)N_NODES * HID * 4);
  float*  agg  = (float*) take((size_t)N_NODES * HID * 4);
  bf16_t* projT = (bf16_t*)take((size_t)HID * KRBF * 2);
  bf16_t* catT  = (bf16_t*)take((size_t)HID * 2 * HID * 2);
  bf16_t* m1T   = (bf16_t*)take((size_t)NLAYERS * HID * KRBF * 2);
  bf16_t* m2T   = (bf16_t*)take((size_t)NLAYERS * HID * HID * 2);
  bf16_t* l1T   = (bf16_t*)take((size_t)NLAYERS * HID * HID * 2);
  bf16_t* l2T   = (bf16_t*)take((size_t)NLAYERS * HID * HID * 2);
  bf16_t* lT    = (bf16_t*)take((size_t)NLAYERS * HID * HID * 2);

  // ---- weight prep: fp32 [K][N] -> bf16 transposed [N][Kpad]
  auto tl = [&](const float* W, bf16_t* WT, int K, int Nm, int Kp) {
    int total = Nm * Kp;
    transpose_bf16_kernel<<<(total + 255) / 256, 256, 0, stream>>>(W, WT, K, Nm, Kp);
  };
  tl(ne_proj_w, projT, NRBF, HID, KRBF);
  tl(ne_cat_w,  catT,  2 * HID, HID, 2 * HID);
  for (int l = 0; l < NLAYERS; ++l) {
    tl(mlp_w1 + (size_t)l * NRBF * HID, m1T + (size_t)l * HID * KRBF, NRBF, HID, KRBF);
    tl(mlp_w2 + (size_t)l * HID * HID,  m2T + (size_t)l * HID * HID,  HID,  HID, HID);
    tl(lin1_w + (size_t)l * HID * HID,  l1T + (size_t)l * HID * HID,  HID,  HID, HID);
    tl(lin2_w + (size_t)l * HID * HID,  l2T + (size_t)l * HID * HID,  HID,  HID, HID);
    tl(lin_w  + (size_t)l * HID * HID,  lT  + (size_t)l * HID * HID,  HID,  HID, HID);
  }

  // ---- per-edge RBF + cutoff
  rbf_kernel<<<(N_EDGES + 255) / 256, 256, 0, stream>>>(ew, rbf_means, rbf_betas, cut, rbfb);

  // ---- embeddings
  embed_kernel<<<(N_NODES * (HID / 4) + 255) / 256, 256, 0, stream>>>(z, emb, ne_emb, x0, y);

  // ---- NeighborEmbedding
  (void)hipMemsetAsync(xn, 0, (size_t)N_NODES * HID * 4, stream);
  const int edgeBlocks = N_EDGES / 16 / 4;   // 4 waves per block, 16 edges per wave
  neighbor_edge_kernel<<<edgeBlocks, 128, 0, stream>>>(rbfb, cut, row, col, y, projT,
                                                       ne_proj_b, xn);
  const int nodeBlocks = N_NODES / 16 / 4;
  cat_kernel<<<nodeBlocks, 128, 0, stream>>>(x0, xn, catT, ne_cat_b, X);

  // ---- interaction layers
  for (int l = 0; l < NLAYERS; ++l) {
    lin1_kernel<<<nodeBlocks, 128, 0, stream>>>(X, l1T + (size_t)l * HID * HID, h);
    (void)hipMemsetAsync(agg, 0, (size_t)N_NODES * HID * 4, stream);
    cfconv_edge_kernel<<<edgeBlocks, 128, 0, stream>>>(
        rbfb, cut, row, col, h,
        m1T + (size_t)l * HID * KRBF, mlp_b1 + (size_t)l * HID,
        m2T + (size_t)l * HID * HID,  mlp_b2 + (size_t)l * HID, agg);
    node_update_kernel<<<nodeBlocks, 128, 0, stream>>>(
        agg, l2T + (size_t)l * HID * HID, lin2_b + (size_t)l * HID,
        lT + (size_t)l * HID * HID, lin_b + (size_t)l * HID, X);
  }
}
